// SlidingWindowProcessor_60593398612039
// MI455X (gfx1250) — matrix operand order repaired and checked by reading, not compiled, not run on hardware
//
#include <hip/hip_runtime.h>
#include <hip/hip_bf16.h>

typedef __attribute__((ext_vector_type(16))) _Float16 v16h;
typedef __attribute__((ext_vector_type(8)))  _Float16 v8h;
typedef __attribute__((ext_vector_type(8)))  float    v8f;

#define Bsz 2
#define Tsz 1024
#define Csz 256
#define Hsz 256
#define Wsz 64
#define Gsz 768

// packed weight sizes (elements)
#define WIH_ELEMS (48*8*32*16)   // 196608  (768x256 -> 48 N-tiles x 8 K-chunks)
#define WPROJ_ELEMS (16*8*32*16) // 65536   (256x256 -> 16 N-tiles x 8 K-chunks)
#define PACK_TOTAL (2*WIH_ELEMS + WPROJ_ELEMS)

#define XPITCH 82   // 41 dwords -> gcd(41,64)=1, conflict-free column writes
#define HPITCH 264  // 132 dwords -> lane rows land on distinct banks for b128 loads

// ---------------------------------------------------------------------------
// Pack w_ih / w_hh / proj_w into WMMA-B fragment-ready f16 layout:
//   out[tile][kk][lane][e] = W[N = 16*tile + lane%16][K = 32*kk + (e%8) + 8*(lane/16) + 16*(e/8)]
// so a B-fragment load is 16 contiguous f16 (32B) per lane, fully coalesced.
// ---------------------------------------------------------------------------
__global__ __launch_bounds__(256) void pack_weights(
    const float* __restrict__ w_ih, const float* __restrict__ w_hh,
    const float* __restrict__ proj_w, _Float16* __restrict__ out)
{
    int idx = blockIdx.x * 256 + threadIdx.x;
    if (idx >= PACK_TOTAL) return;
    const float* src; int i;
    if (idx < WIH_ELEMS)            { src = w_ih;  i = idx; }
    else if (idx < 2 * WIH_ELEMS)   { src = w_hh;  i = idx - WIH_ELEMS; }
    else                            { src = proj_w; i = idx - 2 * WIH_ELEMS; }
    int e    = i & 15;
    int ln   = (i >> 4) & 31;
    int kk   = (i >> 9) & 7;
    int tile = i >> 12;
    int g = tile * 16 + (ln & 15);
    int k = kk * 32 + (e & 7) + ((ln >> 4) << 3) + ((e >> 3) << 4);
    out[idx] = (_Float16)src[g * 256 + k];
}

// ---------------------------------------------------------------------------
// Fused sliding-window GRU + projection + layernorm + GELU.
// One workgroup = 16 consecutive (b,t) sequences. 8 waves; wave w owns gate
// columns [32w, 32w+32): r-tiles {2w,2w+1}, z-tiles {16+2w,17+2w},
// n-tiles {32+2w,33+2w}. 64 sequential GRU steps, WMMA f32_16x16x32_f16.
// h state is ping-ponged between two LDS buffers -> one barrier per step.
// ---------------------------------------------------------------------------
__global__ __launch_bounds__(256) void gru_fused(
    const float* __restrict__ x,
    const float* __restrict__ b_ih, const float* __restrict__ b_hh,
    const float* __restrict__ proj_b,
    const float* __restrict__ ln_g, const float* __restrict__ ln_b,
    const _Float16* __restrict__ wih_p, const _Float16* __restrict__ whh_p,
    const _Float16* __restrict__ wproj_p,
    float* __restrict__ out)
{
    // xwin region is reused for the layernorm epilogue after the GRU loop
    __shared__ __align__(16) char smem_a[Csz * XPITCH * 2];     // 41984 B
    __shared__ __align__(16) _Float16 hbuf[2][16][HPITCH];      // 16896 B

    _Float16 (*xwin)[XPITCH] = (_Float16 (*)[XPITCH])smem_a;

    const int tid  = threadIdx.x;
    const int lane = tid & 31;
    const int wv   = tid >> 5;
    const int m    = lane & 15;     // row inside 16x16 tile / A-matrix M
    const int lhi  = lane >> 4;     // half-wave select
    const int n0   = blockIdx.x * 16;
    const int b    = n0 >> 10;
    const int t0   = n0 & (Tsz - 1);

    // ---- stage shared x window, transposed: xwin[ch][r] = x[b, t0+r-63, ch]
    for (int r = 0; r < 79; ++r) {
        int tt = t0 + r - 63;
        float v = (tt >= 0) ? x[(b * Tsz + tt) * Csz + tid] : 0.0f;
        xwin[tid][r] = (_Float16)v;
    }
    // ---- zero h buffer 0 (input of step 0)
    for (int i = tid; i < 16 * HPITCH; i += 256)
        ((_Float16*)hbuf[0])[i] = (_Float16)0.0f;
    __syncthreads();

    int tiles[6];
    tiles[0] = 2 * wv;      tiles[1] = 2 * wv + 1;
    tiles[2] = 16 + 2 * wv; tiles[3] = 17 + 2 * wv;
    tiles[4] = 32 + 2 * wv; tiles[5] = 33 + 2 * wv;

    float bih[6], bhh[6];
    #pragma unroll
    for (int q = 0; q < 6; ++q) {
        int col = tiles[q] * 16 + m;
        bih[q] = b_ih[col];
        bhh[q] = b_hh[col];
    }

    float hprev[2][8];
    #pragma unroll
    for (int p = 0; p < 2; ++p)
        #pragma unroll
        for (int v = 0; v < 8; ++v) hprev[p][v] = 0.0f;

    for (int step = 0; step < Wsz; ++step) {
        // Opaque zero offset, regenerated every step: defeats LICM (which
        // would hoist all 96 B-fragments out of the loop and spill them to
        // scratch) while keeping the *kernarg* pointer as the load base so
        // address-space inference still emits global_load (not flat_load,
        // which would tick DScnt and couple with the LDS pipeline).
        int zoff = 0;
        asm volatile("" : "+s"(zoff));
        const _Float16* wih_s = wih_p + zoff;
        const _Float16* whh_s = whh_p + zoff;

        const _Float16 (*hin)[HPITCH]  = hbuf[step & 1];
        _Float16 (*hout)[HPITCH]       = hbuf[(step + 1) & 1];

        v8f acc[6], acch[2];
        #pragma unroll
        for (int q = 0; q < 6; ++q) acc[q] = (v8f){};
        acch[0] = (v8f){}; acch[1] = (v8f){};

        // ---- gi = window_row @ w_ih^T  (Hankel gather from xwin)
        #pragma unroll
        for (int kk = 0; kk < 8; ++kk) {
            int ch = 4 * step + (kk >> 1);
            int u0 = (kk & 1) * 32;
            const _Float16* src = &xwin[ch][m + u0 + (lhi << 3)];
            v8h lo, hi;
            __builtin_memcpy(&lo, src, 16);        // K = u0 + 8*lhi + {0..7}
            __builtin_memcpy(&hi, src + 16, 16);   // K = u0 + 8*lhi + {16..23}
            v16h a = __builtin_shufflevector(lo, hi, 0,1,2,3,4,5,6,7,8,9,10,11,12,13,14,15);
            v16h bf[6];
            #pragma unroll
            for (int q = 0; q < 6; ++q)
                bf[q] = *(const v16h*)(wih_s + (((tiles[q] * 8 + kk) * 32 + lane) << 4));
            #pragma unroll
            for (int q = 0; q < 6; ++q)
                acc[q] = __builtin_amdgcn_wmma_f32_16x16x32_f16(
                    false, a, false, bf[q], (short)0, acc[q], false, false);
        }

        // ---- gh = h @ w_hh^T   (aligned b128 A-fragments from hin)
        #pragma unroll
        for (int kk = 0; kk < 8; ++kk) {
            int off0 = 32 * kk + (lhi << 3);
            v8h lo = *(const v8h*)(&hin[m][off0]);
            v8h hi = *(const v8h*)(&hin[m][off0 + 16]);
            v16h a = __builtin_shufflevector(lo, hi, 0,1,2,3,4,5,6,7,8,9,10,11,12,13,14,15);
            v16h bf[6];
            #pragma unroll
            for (int q = 0; q < 6; ++q)
                bf[q] = *(const v16h*)(whh_s + (((tiles[q] * 8 + kk) * 32 + lane) << 4));
            #pragma unroll
            for (int q = 0; q < 4; ++q)   // r,z gates: accumulate on top of gi
                acc[q] = __builtin_amdgcn_wmma_f32_16x16x32_f16(
                    false, a, false, bf[q], (short)0, acc[q], false, false);
            #pragma unroll
            for (int q = 4; q < 6; ++q)   // n gate: keep h-part separate (r * h_n)
                acch[q - 4] = __builtin_amdgcn_wmma_f32_16x16x32_f16(
                    false, a, false, bf[q], (short)0, acch[q - 4], false, false);
        }

        // ---- GRU elementwise update (wave-local columns) -> hout
        #pragma unroll
        for (int p = 0; p < 2; ++p) {
            #pragma unroll
            for (int v = 0; v < 8; ++v) {
                float rg = acc[p][v]     + bih[p]     + bhh[p];
                float zg = acc[2 + p][v] + bih[2 + p] + bhh[2 + p];
                float r  = 1.0f / (1.0f + expf(-rg));
                float z  = 1.0f / (1.0f + expf(-zg));
                float ng = acc[4 + p][v] + bih[4 + p] + r * (acch[p][v] + bhh[4 + p]);
                float nn = tanhf(ng);
                float h  = (1.0f - z) * nn + z * hprev[p][v];
                hprev[p][v] = h;
                hout[v + 8 * lhi][32 * wv + 16 * p + m] = (_Float16)h;
            }
        }
        __syncthreads();   // hout complete before next step reads it
    }

    // final h state lives in hbuf[0] (input of "step 64")
    const _Float16 (*hfin)[HPITCH] = hbuf[0];

    // ---- projection: y = hT @ proj_w^T + proj_b   (2 tiles per wave)
    float (*ybuf)[Hsz] = (float (*)[Hsz])smem_a;                 // 16 KB
    float* redsum = (float*)(smem_a + 16 * Hsz * 4);             // 16x16
    float* redsq  = redsum + 256;                                // 16x16
    float* stats  = redsq + 256;                                 // 16x2
    {
        v8f acc2[2]; acc2[0] = (v8f){}; acc2[1] = (v8f){};
        int pt0 = 2 * wv, pt1 = 2 * wv + 1;
        #pragma unroll
        for (int kk = 0; kk < 8; ++kk) {
            int off0 = 32 * kk + (lhi << 3);
            v8h lo = *(const v8h*)(&hfin[m][off0]);
            v8h hi = *(const v8h*)(&hfin[m][off0 + 16]);
            v16h a = __builtin_shufflevector(lo, hi, 0,1,2,3,4,5,6,7,8,9,10,11,12,13,14,15);
            v16h bf0 = *(const v16h*)(wproj_p + (((pt0 * 8 + kk) * 32 + lane) << 4));
            v16h bf1 = *(const v16h*)(wproj_p + (((pt1 * 8 + kk) * 32 + lane) << 4));
            acc2[0] = __builtin_amdgcn_wmma_f32_16x16x32_f16(
                false, a, false, bf0, (short)0, acc2[0], false, false);
            acc2[1] = __builtin_amdgcn_wmma_f32_16x16x32_f16(
                false, a, false, bf1, (short)0, acc2[1], false, false);
        }
        __syncthreads();   // done with xwin region; safe to reuse as ybuf
        float pb0 = proj_b[pt0 * 16 + m];
        float pb1 = proj_b[pt1 * 16 + m];
        #pragma unroll
        for (int v = 0; v < 8; ++v) {
            ybuf[v + 8 * lhi][pt0 * 16 + m] = acc2[0][v] + pb0;
            ybuf[v + 8 * lhi][pt1 * 16 + m] = acc2[1][v] + pb1;
        }
    }
    __syncthreads();

    // ---- layernorm stats (16 rows, 16 partials each)
    {
        int row = tid >> 4, seg = tid & 15;
        float s = 0.0f, sq = 0.0f;
        #pragma unroll
        for (int i = 0; i < 16; ++i) {
            float v = ybuf[row][seg * 16 + i];
            s += v; sq += v * v;
        }
        redsum[row * 16 + seg] = s;
        redsq[row * 16 + seg]  = sq;
    }
    __syncthreads();
    if (tid < 16) {
        float s = 0.0f, sq = 0.0f;
        for (int i = 0; i < 16; ++i) { s += redsum[tid * 16 + i]; sq += redsq[tid * 16 + i]; }
        float mu  = s * (1.0f / 256.0f);
        float var = sq * (1.0f / 256.0f) - mu * mu;
        stats[tid * 2 + 0] = mu;
        stats[tid * 2 + 1] = rsqrtf(var + 1e-5f);
    }
    __syncthreads();

    // ---- normalize + exact GELU, coalesced store
    {
        float lg = ln_g[tid & 255], lb = ln_b[tid & 255];
        for (int row = 0; row < 16; ++row) {
            float yn = (ybuf[row][tid] - stats[row * 2]) * stats[row * 2 + 1] * lg + lb;
            float o  = 0.5f * yn * (1.0f + erff(yn * 0.70710678118654752f));
            out[(n0 + row) * Hsz + tid] = o;
        }
    }
}

extern "C" void kernel_launch(void* const* d_in, const int* in_sizes, int n_in,
                              void* d_out, int out_size, void* d_ws, size_t ws_size,
                              hipStream_t stream) {
    const float* x      = (const float*)d_in[0];
    const float* w_ih   = (const float*)d_in[1];
    const float* w_hh   = (const float*)d_in[2];
    const float* b_ih   = (const float*)d_in[3];
    const float* b_hh   = (const float*)d_in[4];
    const float* proj_w = (const float*)d_in[5];
    const float* proj_b = (const float*)d_in[6];
    const float* ln_g   = (const float*)d_in[7];
    const float* ln_b   = (const float*)d_in[8];
    float* outp = (float*)d_out;

    _Float16* wih_p   = (_Float16*)d_ws;
    _Float16* whh_p   = wih_p + WIH_ELEMS;
    _Float16* wproj_p = whh_p + WIH_ELEMS;

    // pack weights (re-done every call; ws may be poisoned)
    int pack_blocks = (PACK_TOTAL + 255) / 256;
    pack_weights<<<pack_blocks, 256, 0, stream>>>(w_ih, w_hh, proj_w, wih_p);

    // fused GRU pipeline: 2048 sequences / 16 per workgroup
    gru_fused<<<128, 256, 0, stream>>>(x, b_ih, b_hh, proj_b, ln_g, ln_b,
                                       wih_p, whh_p, wproj_p, outp);
}
